// MaskedGenerativeEncoderViT_14516989460899
// MI455X (gfx1250) — matrix operand order repaired and checked by reading, not compiled, not run on hardware
//
// MaskedGenerativeEncoderViT forward for MI455X (gfx1250), wave32 + WMMA bf16.
//
// Roofline: ~1.0 TFLOP of GEMM. Weights (~0.25 GB fp32) are converted to bf16
// ONCE per launch (~124 MB, fits the 192 MB L2), and every GEMM operand is
// produced in bf16 by the upstream kernel, so the WMMA K-loop is pure
// b128 loads + v_wmma (no cvt, no EXEC juggling). The loop is software-
// pipelined (double-buffered fragments) so each wave's WMMAs consume data
// loaded one full iteration earlier. fp32 residual state + fp32 accumulate
// preserve accuracy. Attention (<4% of FLOPs) is flash-style VALU.
#include <hip/hip_runtime.h>
#include <hip/hip_bf16.h>
#include <cmath>

typedef __bf16 bf16_t;
typedef __attribute__((ext_vector_type(16))) bf16_t v16bf;
typedef __attribute__((ext_vector_type(8)))  bf16_t v8bf;
typedef __attribute__((ext_vector_type(4)))  bf16_t v4bf;
typedef __attribute__((ext_vector_type(8)))  float  v8f;

// ---------------------------------------------------------------- utilities
__device__ __forceinline__ float gelu_exact(float x) {
  return 0.5f * x * (1.0f + erff(x * 0.70710678118654752f));
}

// Full block reduction of (sum, sumsq); result broadcast to every lane.
__device__ __forceinline__ void block_reduce_sum2(float& s, float& s2) {
  #pragma unroll
  for (int o = 16; o; o >>= 1) {
    s  += __shfl_xor(s,  o, 32);
    s2 += __shfl_xor(s2, o, 32);
  }
  __shared__ float sh0[8], sh1[8];
  const int wave = threadIdx.x >> 5;
  const int lane = threadIdx.x & 31;
  const int nw   = blockDim.x >> 5;
  if (lane == 0) { sh0[wave] = s; sh1[wave] = s2; }
  __syncthreads();
  float a = (lane < nw) ? sh0[lane] : 0.f;
  float b = (lane < nw) ? sh1[lane] : 0.f;
  #pragma unroll
  for (int o = 16; o; o >>= 1) {
    a += __shfl_xor(a, o, 32);
    b += __shfl_xor(b, o, 32);
  }
  s = a; s2 = b;
}

// -------------------------------------------------------- fp32 -> bf16 copy
__global__ __launch_bounds__(256) void f32_to_bf16_k(
    const float* __restrict__ in, bf16_t* __restrict__ out, long long n) {
  const long long i = ((long long)blockIdx.x * 256 + threadIdx.x) * 4;
  if (i >= n) return;                       // n is always a multiple of 4
  float4 f = *reinterpret_cast<const float4*>(in + i);
  v4bf o;
  o[0] = (bf16_t)f.x; o[1] = (bf16_t)f.y; o[2] = (bf16_t)f.z; o[3] = (bf16_t)f.w;
  *reinterpret_cast<v4bf*>(out + i) = o;
}

// ------------------------------------------------------------- embed + LN
// X[m] = LN(word_emb[ids[m]] + pos_emb[m % Ntok])   (fp32 residual state)
__global__ __launch_bounds__(256) void embed_ln_k(
    const int* __restrict__ ids, const float* __restrict__ wemb,
    const float* __restrict__ pemb, const float* __restrict__ g,
    const float* __restrict__ bb, float* __restrict__ out,
    int Ntok, int E, float eps) {
  const int m   = blockIdx.x;
  const int tok = m % Ntok;
  const long long id = ids[m];
  const float* we = wemb + id * (long long)E;
  const float* pe = pemb + (long long)tok * E;
  float vals[4];
  float s = 0.f, s2 = 0.f;
  #pragma unroll
  for (int j = 0; j < 4; ++j) {
    int i = threadIdx.x + j * 256;
    float v = we[i] + pe[i];
    vals[j] = v; s += v; s2 += v * v;
  }
  block_reduce_sum2(s, s2);
  const float mean = s / E;
  const float rinv = rsqrtf(s2 / E - mean * mean + eps);
  float* o = out + (long long)m * E;
  #pragma unroll
  for (int j = 0; j < 4; ++j) {
    int i = threadIdx.x + j * 256;
    o[i] = (vals[j] - mean) * rinv * g[i] + bb[i];
  }
}

// ------------------------------------------- LayerNorm: fp32 in, bf16 out
__global__ __launch_bounds__(256) void layernorm_k(
    const float* __restrict__ in, const float* __restrict__ g,
    const float* __restrict__ bb, bf16_t* __restrict__ out, int C, float eps) {
  const int row = blockIdx.x;
  const float* x = in + (long long)row * C;
  float s = 0.f, s2 = 0.f;
  for (int i = threadIdx.x; i < C; i += blockDim.x) {
    float v = x[i]; s += v; s2 += v * v;
  }
  block_reduce_sum2(s, s2);
  const float mean = s / C;
  const float rinv = rsqrtf(s2 / C - mean * mean + eps);
  bf16_t* o = out + (long long)row * C;
  for (int i = threadIdx.x; i < C; i += blockDim.x)
    o[i] = (bf16_t)((x[i] - mean) * rinv * g[i] + bb[i]);
}

// ------------------------------------------------------ WMMA bf16 GEMM
// epilogue(A @ W^T + bias); A:(M,K) bf16, W:(Nout,K) bf16 row-major (B=W^T).
// Contiguous k order (k = 16*(lane/16)+s) in A and B. Wave tile 16x64 (1 A frag -> 4 WMMA), software-
// pipelined: loads for k+32 issue before the WMMAs for k. Out-of-range N
// columns are CLAMPED for loads (branch-free loop) and guarded at store.
// mode 0: +bias | 1: +bias+resid(M,Nout) | 2: gelu(+bias) | 3: +bias+pos(row%posN)
// outh!=null -> bf16 output, else fp32 to outf.
__global__ __launch_bounds__(256) void gemm_bf16_wmma(
    const bf16_t* __restrict__ A, const bf16_t* __restrict__ W,
    const float* __restrict__ bias, const float* __restrict__ resid,
    float* __restrict__ outf, bf16_t* __restrict__ outh,
    int M, int K, int Nout, int mode, int posN) {
  const int lane = threadIdx.x & 31;
  const int wave = threadIdx.x >> 5;
  const int ln15 = lane & 15;
  const int hi   = lane >> 4;
  const int M0 = blockIdx.y * 32 + (wave >> 2) * 16;   // M is a multiple of 32
  const int N0 = blockIdx.x * 256 + (wave & 3) * 64;

  // A: lanes 0-15 row M0+lane K{0..15}; lanes 16-31 same rows K{16..31}
  const bf16_t* ap = A + (long long)(M0 + ln15) * K + (hi << 4);
  // B: lanes 0-15 N=col K k0+0..15; lanes 16-31 same N K k0+16..31 (clamped col)
  long long woff[4];
  #pragma unroll
  for (int s = 0; s < 4; ++s) {
    int c = N0 + s * 16 + ln15;
    c = (c < Nout) ? c : (Nout - 1);
    woff[s] = (long long)c * K + (hi << 4);
  }

  auto loadA = [&](int k0) -> v16bf {
    v8bf lo = *reinterpret_cast<const v8bf*>(ap + k0);
    v8bf hb = *reinterpret_cast<const v8bf*>(ap + k0 + 8);
    return __builtin_shufflevector(lo, hb,
        0, 1, 2, 3, 4, 5, 6, 7, 8, 9, 10, 11, 12, 13, 14, 15);
  };
  auto loadB = [&](int s, int k0) -> v16bf {
    return *reinterpret_cast<const v16bf*>(W + woff[s] + k0);
  };

  v8f acc[4] = {v8f{}, v8f{}, v8f{}, v8f{}};
  v16bf a  = loadA(0);
  v16bf b0 = loadB(0, 0), b1 = loadB(1, 0), b2 = loadB(2, 0), b3 = loadB(3, 0);
  for (int k0 = 0; k0 < K; k0 += 32) {
    // issue next iteration's loads first (double buffer; last iter reloads k0)
    const int kn = (k0 + 32 < K) ? (k0 + 32) : k0;
    v16bf an  = loadA(kn);
    v16bf bn0 = loadB(0, kn);
    v16bf bn1 = loadB(1, kn);
    v16bf bn2 = loadB(2, kn);
    v16bf bn3 = loadB(3, kn);
    __builtin_prefetch(ap + kn + 32, 0, 3);      // global_prefetch_b8
    acc[0] = __builtin_amdgcn_wmma_f32_16x16x32_bf16(false, a, false, b0,
                                                     (short)0, acc[0], false, false);
    acc[1] = __builtin_amdgcn_wmma_f32_16x16x32_bf16(false, a, false, b1,
                                                     (short)0, acc[1], false, false);
    acc[2] = __builtin_amdgcn_wmma_f32_16x16x32_bf16(false, a, false, b2,
                                                     (short)0, acc[2], false, false);
    acc[3] = __builtin_amdgcn_wmma_f32_16x16x32_bf16(false, a, false, b3,
                                                     (short)0, acc[3], false, false);
    a = an; b0 = bn0; b1 = bn1; b2 = bn2; b3 = bn3;
  }

  // C/D: VGPR v -> M = v (+8 for lanes 16-31); N = lane&15
  const int row_base = M0 + (hi << 3);
  #pragma unroll
  for (int s = 0; s < 4; ++s) {
    const int col = N0 + s * 16 + ln15;
    if (col >= Nout) continue;
    const float bv = bias ? bias[col] : 0.0f;
    #pragma unroll
    for (int v = 0; v < 8; ++v) {
      const int row = row_base + v;
      float val = acc[s][v] + bv;
      if (mode == 2)      val = gelu_exact(val);
      else if (mode == 1) val += resid[(long long)row * Nout + col];
      else if (mode == 3) val += resid[(long long)(row % posN) * Nout + col];
      if (outh) outh[(long long)row * Nout + col] = (bf16_t)val;
      else      outf[(long long)row * Nout + col] = val;
    }
  }
}

// ------------------------------------------- flash-style attention (VALU)
// One wave per query row; lanes parallel over dh (dh in {32,64}).
// Reads fp32 Q/K/V, writes bf16 (consumed only by the next GEMM).
__global__ __launch_bounds__(256) void attn_flash_k(
    const float* __restrict__ Q, const float* __restrict__ K,
    const float* __restrict__ V, bf16_t* __restrict__ O,
    int Tq, int Tk, int nh, int dh,
    long long qTok, long long qBat, long long kTok, long long kBat,
    long long oTok, long long oBat, float scale) {
  const int lane = threadIdx.x & 31;
  const int wave = threadIdx.x >> 5;
  const int b = blockIdx.x / nh;
  const int h = blockIdx.x % nh;
  const int qr = blockIdx.y * 8 + wave;
  if (qr >= Tq) return;                       // wave-uniform
  const int D = dh >> 5;                      // 1 or 2 values per lane
  const float* qp = Q + (long long)b * qBat + (long long)qr * qTok + h * dh;
  float qv[2] = {0.f, 0.f}, av[2] = {0.f, 0.f};
  for (int j = 0; j < D; ++j) qv[j] = qp[lane + 32 * j] * scale;
  const float* kb = K + (long long)b * kBat + h * dh;
  const float* vb = V + (long long)b * kBat + h * dh;
  float mx = -3.0e38f, l = 0.f;
  for (int t = 0; t < Tk; ++t) {
    const float* kp = kb + (long long)t * kTok;
    float s = 0.f;
    for (int j = 0; j < D; ++j) s += qv[j] * kp[lane + 32 * j];
    #pragma unroll
    for (int o = 16; o; o >>= 1) s += __shfl_xor(s, o, 32);
    const float mn = fmaxf(mx, s);
    const float cf = __expf(mx - mn);
    const float p  = __expf(s - mn);
    l = l * cf + p;
    const float* vp = vb + (long long)t * kTok;
    for (int j = 0; j < D; ++j) av[j] = av[j] * cf + p * vp[lane + 32 * j];
    mx = mn;
  }
  const float rl = 1.0f / l;
  bf16_t* op = O + (long long)b * oBat + (long long)qr * oTok + h * dh;
  for (int j = 0; j < D; ++j) op[lane + 32 * j] = (bf16_t)(av[j] * rl);
}

// ---------------------------------------------------------------- host
static inline int ceil_div(int a, int b) { return (a + b - 1) / b; }

static void run_gemm(hipStream_t st, const bf16_t* A, const bf16_t* W,
                     const float* bias, const float* resid,
                     float* outf, bf16_t* outh,
                     int M, int K, int Nout, int mode, int posN = 1) {
  dim3 grid(ceil_div(Nout, 256), M / 32);
  gemm_bf16_wmma<<<grid, 256, 0, st>>>(A, W, bias, resid, outf, outh,
                                       M, K, Nout, mode, posN);
}

extern "C" void kernel_launch(void* const* d_in, const int* in_sizes, int n_in,
                              void* d_out, int out_size, void* d_ws, size_t ws_size,
                              hipStream_t stream) {
  (void)in_sizes; (void)n_in; (void)out_size; (void)ws_size;
  const int B_ = 32, Ntok = 257, E = 1024, NH = 16, DEPTH = 4;
  const int DD = 512, NHD = 16, DDEPTH = 2, VOCAB = 2025, NS = 7;
  const int M  = B_ * Ntok;      // 8224 = 257 * 32
  const int MS = B_ * NS;        // 224  = 7 * 32

  const int*   ids   = (const int*)d_in[0];
  const float* slots = (const float*)d_in[1];
  auto F = [&](int i) { return (const float*)d_in[i]; };

  // params subtree flattened jax-style (sorted dict keys), after ids/slots.
  const int DEC0 = 2;   // 2 blocks x 18 sorted leaves
  enum { D_FC1_B = 0, D_FC1_W, D_FC2_B, D_FC2_W, D_LN1_B, D_LN1_G, D_LN2_B,
         D_LN2_G, D_LNC_B, D_LNC_G, D_PROJ_B, D_PROJ_W, D_QKV_B, D_QKV_W,
         D_WK, D_WO, D_WQ, D_WV };
  auto dec = [&](int blk, int w) { return F(DEC0 + 18 * blk + w); };
  const float* dec_embed_b = F(38);
  const float* dec_embed_w = F(39);
  const float* dec_ln_b    = F(40);
  const float* dec_ln_g    = F(41);
  const float* dec_pos     = F(42);
  const float* emb_ln_b    = F(43);
  const float* emb_ln_g    = F(44);
  const int ENC0 = 45;  // 4 blocks x 12 sorted leaves
  enum { E_FC1_B = 0, E_FC1_W, E_FC2_B, E_FC2_W, E_LN1_B, E_LN1_G, E_LN2_B,
         E_LN2_G, E_PROJ_B, E_PROJ_W, E_QKV_B, E_QKV_W };
  auto enc = [&](int blk, int w) { return F(ENC0 + 12 * blk + w); };
  const float* enc_ln_b = F(93);
  const float* enc_ln_g = F(94);
  const float* mlm_bias = F(95);
  const float* mlm_fc_b = F(96);
  const float* mlm_fc_w = F(97);
  const float* mlm_ln_b = F(98);
  const float* mlm_ln_g = F(99);
  const float* pos_emb  = F(100);
  const float* word_emb = F(101);

  // ---- workspace carve-out: fp32 state/attn buffers + bf16 operand arena
  float* X  = (float*)d_ws;                  // M x 1024 : residual / mlm hidden
  float* Cb = X  + (size_t)M * 1024;         // M x 3072 : qkv (fp32, attn input)
  float* Qb = Cb + (size_t)M * 3072;         // M x 512  : cross-attn Q
  float* Kb = Qb + (size_t)M * 512;          // 224 x 512
  float* Vb = Kb + (size_t)MS * 512;         // 224 x 512
  bf16_t* Ybf = (bf16_t*)(Vb + (size_t)MS * 512); // M x 1024 : LN / attn out
  bf16_t* Hbf = Ybf + (size_t)M * 1024;      // M x 4096 : mlp hidden (bf16)
  bf16_t* arena = Hbf + (size_t)M * 4096;    // bf16 weights (~124 MB)

  // ---- one-time fp32 -> bf16 conversion of all GEMM operands that are inputs
  bf16_t* wp = arena;
  auto cvt = [&](const float* src, long long n) -> const bf16_t* {
    bf16_t* dst = wp;
    wp += (n + 15) & ~15LL;
    long long blocks = (n / 4 + 255) / 256;
    f32_to_bf16_k<<<dim3((unsigned)blocks), 256, 0, stream>>>(src, dst, n);
    return dst;
  };
  const bf16_t *eqkv[4], *eproj[4], *efc1[4], *efc2[4];
  for (int i = 0; i < DEPTH; ++i) {
    eqkv[i]  = cvt(enc(i, E_QKV_W),  3LL * E * E);
    eproj[i] = cvt(enc(i, E_PROJ_W), (long long)E * E);
    efc1[i]  = cvt(enc(i, E_FC1_W),  4LL * E * E);
    efc2[i]  = cvt(enc(i, E_FC2_W),  4LL * E * E);
  }
  const bf16_t *dqkv[2], *dproj[2], *dfc1[2], *dfc2[2], *dwq[2], *dwk[2], *dwv[2], *dwo[2];
  for (int i = 0; i < DDEPTH; ++i) {
    dqkv[i]  = cvt(dec(i, D_QKV_W),  3LL * DD * DD);
    dproj[i] = cvt(dec(i, D_PROJ_W), (long long)DD * DD);
    dfc1[i]  = cvt(dec(i, D_FC1_W),  4LL * DD * DD);
    dfc2[i]  = cvt(dec(i, D_FC2_W),  4LL * DD * DD);
    dwq[i]   = cvt(dec(i, D_WQ),     (long long)DD * DD);
    dwk[i]   = cvt(dec(i, D_WK),     (long long)DD * DD);
    dwv[i]   = cvt(dec(i, D_WV),     (long long)DD * DD);
    dwo[i]   = cvt(dec(i, D_WO),     (long long)DD * DD);
  }
  const bf16_t* wemb_bf  = cvt(word_emb,    (long long)VOCAB * E);
  const bf16_t* demb_bf  = cvt(dec_embed_w, (long long)DD * E);
  const bf16_t* mlmfc_bf = cvt(mlm_fc_w,    (long long)E * DD);
  const bf16_t* slots_bf = cvt(slots,       (long long)MS * DD);

  // ---- embeddings + LN(1e-6) -> fp32 state X
  embed_ln_k<<<M, 256, 0, stream>>>(ids, word_emb, pos_emb, emb_ln_g, emb_ln_b,
                                    X, Ntok, E, 1e-6f);

  // ---- encoder: x += attn(ln1 x); x += mlp(ln2 x)
  const int dhE = E / NH;                                  // 64
  const float scE = 1.0f / sqrtf((float)dhE);
  for (int i = 0; i < DEPTH; ++i) {
    layernorm_k<<<M, 256, 0, stream>>>(X, enc(i, E_LN1_G), enc(i, E_LN1_B), Ybf, E, 1e-5f);
    run_gemm(stream, Ybf, eqkv[i], enc(i, E_QKV_B), nullptr, Cb, nullptr, M, E, 3 * E, 0);
    {
      dim3 g(B_ * NH, ceil_div(Ntok, 8));
      attn_flash_k<<<g, 256, 0, stream>>>(Cb, Cb + E, Cb + 2 * E, Ybf,
          Ntok, Ntok, NH, dhE,
          3LL * E, (long long)Ntok * 3 * E, 3LL * E, (long long)Ntok * 3 * E,
          (long long)E, (long long)Ntok * E, scE);
    }
    run_gemm(stream, Ybf, eproj[i], enc(i, E_PROJ_B), X, X, nullptr, M, E, E, 1);
    layernorm_k<<<M, 256, 0, stream>>>(X, enc(i, E_LN2_G), enc(i, E_LN2_B), Ybf, E, 1e-5f);
    run_gemm(stream, Ybf, efc1[i], enc(i, E_FC1_B), nullptr, nullptr, Hbf, M, E, 4 * E, 2);
    run_gemm(stream, Hbf, efc2[i], enc(i, E_FC2_B), X, X, nullptr, M, 4 * E, E, 1);
  }
  layernorm_k<<<M, 256, 0, stream>>>(X, enc_ln_g, enc_ln_b, Ybf, E, 1e-5f);

  // ---- decoder embed: X = Ybf @ dec_embed_w^T + b + dec_pos[token]
  run_gemm(stream, Ybf, demb_bf, dec_embed_b, dec_pos, X, nullptr, M, E, DD, 3, Ntok);

  // ---- decoder: self-attn + cross-attn(slots) + mlp
  const int dhD = DD / NHD;                                // 32
  const float scD = 1.0f / sqrtf((float)dhD);
  for (int i = 0; i < DDEPTH; ++i) {
    layernorm_k<<<M, 256, 0, stream>>>(X, dec(i, D_LN1_G), dec(i, D_LN1_B), Ybf, DD, 1e-5f);
    run_gemm(stream, Ybf, dqkv[i], dec(i, D_QKV_B), nullptr, Cb, nullptr, M, DD, 3 * DD, 0);
    {
      dim3 g(B_ * NHD, ceil_div(Ntok, 8));
      attn_flash_k<<<g, 256, 0, stream>>>(Cb, Cb + DD, Cb + 2 * DD, Ybf,
          Ntok, Ntok, NHD, dhD,
          3LL * DD, (long long)Ntok * 3 * DD, 3LL * DD, (long long)Ntok * 3 * DD,
          (long long)DD, (long long)Ntok * DD, scD);
    }
    run_gemm(stream, Ybf, dproj[i], dec(i, D_PROJ_B), X, X, nullptr, M, DD, DD, 1);

    // cross-attention to slots
    layernorm_k<<<M, 256, 0, stream>>>(X, dec(i, D_LNC_G), dec(i, D_LNC_B), Ybf, DD, 1e-5f);
    run_gemm(stream, Ybf,      dwq[i], nullptr, nullptr, Qb, nullptr, M,  DD, DD, 0);
    run_gemm(stream, slots_bf, dwk[i], nullptr, nullptr, Kb, nullptr, MS, DD, DD, 0);
    run_gemm(stream, slots_bf, dwv[i], nullptr, nullptr, Vb, nullptr, MS, DD, DD, 0);
    {
      dim3 g(B_ * NHD, ceil_div(Ntok, 8));
      attn_flash_k<<<g, 256, 0, stream>>>(Qb, Kb, Vb, Ybf,
          Ntok, NS, NHD, dhD,
          (long long)DD, (long long)Ntok * DD, (long long)DD, (long long)NS * DD,
          (long long)DD, (long long)Ntok * DD, scD);
    }
    run_gemm(stream, Ybf, dwo[i], nullptr, X, X, nullptr, M, DD, DD, 1);

    layernorm_k<<<M, 256, 0, stream>>>(X, dec(i, D_LN2_G), dec(i, D_LN2_B), Ybf, DD, 1e-5f);
    run_gemm(stream, Ybf, dfc1[i], dec(i, D_FC1_B), nullptr, nullptr, Hbf, M, DD, 4 * DD, 2);
    run_gemm(stream, Hbf, dfc2[i], dec(i, D_FC2_B), X, X, nullptr, M, 4 * DD, DD, 1);
  }
  layernorm_k<<<M, 256, 0, stream>>>(X, dec_ln_g, dec_ln_b, Ybf, DD, 1e-5f);

  // ---- MLM head: gelu(fc) -> LN -> tied-embedding logits + bias
  run_gemm(stream, Ybf, mlmfc_bf, mlm_fc_b, nullptr, X, nullptr, M, DD, E, 2);
  layernorm_k<<<M, 256, 0, stream>>>(X, mlm_ln_g, mlm_ln_b, Ybf, E, 1e-5f);
  run_gemm(stream, Ybf, wemb_bf, mlm_bias, nullptr, (float*)d_out, nullptr, M, E, VOCAB, 0);
}
